// BiLSTM_CRF_23708219474277
// MI455X (gfx1250) — matrix operand order repaired and checked by reading, not compiled, not run on hardware
//
#include <hip/hip_runtime.h>
#include <hip/hip_bf16.h>
#include <math.h>

// ---------------------------------------------------------------------------
// BiLSTM-CRF for MI455X (gfx1250, wave32).
//   - Input projections X @ Wih^T run on WMMA (v_wmma_f32_16x16x32_bf16).
//   - LSTM recurrence: persistent 1024-thread workgroup per direction,
//     h/c/gates in LDS, serialized over T with workgroup barriers.
//   - Viterbi: single-wave serial scan + backtrack.
// ---------------------------------------------------------------------------

#define T_LEN   4096
#define E_DIM   256
#define P_DIM   64
#define D_DIM   320      // LSTM input size (E+P), 10 x 32 -> clean WMMA K
#define HH_DIM  256      // hidden per direction
#define H_DIM   512
#define G_DIM   1024     // 4*HH gates
#define NTAGS   12
#define START_TAG 10
#define STOP_TAG  11
#define NEG_VAL  -10000.0f

typedef __bf16 bf16_t;
typedef __attribute__((ext_vector_type(8)))  __bf16 v8bf;
typedef __attribute__((ext_vector_type(16))) __bf16 v16bf;
typedef __attribute__((ext_vector_type(8)))  float  v8f;

// ----- workspace layout (bytes, 256-aligned) -------------------------------
static constexpr size_t OFF_XB    = 0;                         // 4096*320 bf16
static constexpr size_t OFF_WBF   = OFF_XB    + (size_t)T_LEN * D_DIM * 2;   // 1024*320 bf16
static constexpr size_t OFF_WBB   = OFF_WBF   + (size_t)G_DIM * D_DIM * 2;
static constexpr size_t OFF_PREF  = OFF_WBB   + (size_t)G_DIM * D_DIM * 2;   // 4096*1024 f32
static constexpr size_t OFF_PREB  = OFF_PREF  + (size_t)T_LEN * G_DIM * 4;
static constexpr size_t OFF_HF    = OFF_PREB  + (size_t)T_LEN * G_DIM * 4;   // 4096*256 f32
static constexpr size_t OFF_HB    = OFF_HF    + (size_t)T_LEN * HH_DIM * 4;
static constexpr size_t OFF_FEATS = OFF_HB    + (size_t)T_LEN * HH_DIM * 4;  // 4096*12 f32
static constexpr size_t OFF_BP    = OFF_FEATS + (size_t)T_LEN * NTAGS * 4;   // 4096*12 i32

__device__ __forceinline__ float sigmoidf_fast(float x) {
    return 1.0f / (1.0f + __expf(-x));
}

// ---------------------------------------------------------------------------
// 1) Embedding gather -> bf16 X (T x D)
// ---------------------------------------------------------------------------
__global__ void embed_kernel(const int* __restrict__ words,
                             const int* __restrict__ pos,
                             const float* __restrict__ Wword,
                             const float* __restrict__ Wpos,
                             bf16_t* __restrict__ Xb) {
    int idx = blockIdx.x * blockDim.x + threadIdx.x;
    if (idx >= T_LEN * D_DIM) return;
    int t = idx / D_DIM;
    int k = idx - t * D_DIM;
    float v;
    if (k < E_DIM) v = Wword[(size_t)words[t] * E_DIM + k];
    else           v = Wpos[(size_t)pos[t] * P_DIM + (k - E_DIM)];
    Xb[(size_t)t * D_DIM + k] = (bf16_t)v;
}

// ---------------------------------------------------------------------------
// 2) Wih (both directions) f32 -> bf16
// ---------------------------------------------------------------------------
__global__ void wcvt_kernel(const float* __restrict__ WihF,
                            const float* __restrict__ WihB,
                            bf16_t* __restrict__ WbF,
                            bf16_t* __restrict__ WbB) {
    int idx = blockIdx.x * blockDim.x + threadIdx.x;
    const int N = G_DIM * D_DIM;
    if (idx >= 2 * N) return;
    if (idx < N) WbF[idx] = (bf16_t)WihF[idx];
    else         WbB[idx - N] = (bf16_t)WihB[idx - N];
}

// ---------------------------------------------------------------------------
// 3) WMMA GEMM: pre[dir] = X (4096x320) @ W[dir]^T (320x1024)
//    grid = (256 Mtiles, 8, 2 dirs), 256 threads = 8 waves,
//    one 16x16 output tile per wave, K swept in 10 steps of 32.
//    Operand layouts per CDNA5 ISA 7.12.2 (16-bit A 16x32 / B 32x16).
// ---------------------------------------------------------------------------
__global__ __launch_bounds__(256)
void gemm_kernel(const bf16_t* __restrict__ Xb,
                 const bf16_t* __restrict__ WbF,
                 const bf16_t* __restrict__ WbB,
                 float* __restrict__ preF,
                 float* __restrict__ preB) {
    const int lane  = threadIdx.x & 31;
    const int wave  = threadIdx.x >> 5;
    const int mtile = blockIdx.x;                 // 0..255
    const int ntile = blockIdx.y * 8 + wave;      // 0..63
    const int dir   = blockIdx.z;

    const bf16_t* __restrict__ W   = dir ? WbB  : WbF;
    float*        __restrict__ pre = dir ? preB : preF;

    const int half = lane >> 4;                   // 0 | 1
    const int l15  = lane & 15;
    const int arow = mtile * 16 + l15;            // A row (M)
    const int koffA = half * 8;                   // A: lanes 16-31 carry K+8
    const int bcol = ntile * 16 + l15;            // B col (N) == gate index
    const int kbB  = half * 8;                    // B: lanes 16-31 carry K+8

    v8f acc = {};
    const bf16_t* arowp = Xb + (size_t)arow * D_DIM;
    const bf16_t* bcolp = W  + (size_t)bcol * D_DIM;   // W[g][k] contiguous in k

#pragma unroll
    for (int k0 = 0; k0 < D_DIM; k0 += 32) {
        v8bf a_lo = *(const v8bf*)(arowp + k0 + koffA);
        v8bf a_hi = *(const v8bf*)(arowp + k0 + 16 + koffA);
        v8bf b_lo = *(const v8bf*)(bcolp + k0 + kbB);
        v8bf b_hi = *(const v8bf*)(bcolp + k0 + 16 + kbB);
        v16bf a, b;
#pragma unroll
        for (int i = 0; i < 8; ++i) {
            a[i] = a_lo[i]; a[8 + i] = a_hi[i];
            b[i] = b_lo[i]; b[8 + i] = b_hi[i];
        }
        acc = __builtin_amdgcn_wmma_f32_16x16x32_bf16(
            /*neg_a=*/false, a, /*neg_b=*/false, b,
            /*c_mod=*/(short)0, acc, /*reuse_a=*/false, /*reuse_b=*/false);
    }

#pragma unroll
    for (int r = 0; r < 8; ++r) {
        int orow = mtile * 16 + half * 8 + r;     // D layout: M = vgpr + half*8
        pre[(size_t)orow * G_DIM + bcol] = acc[r];
    }
}

// ---------------------------------------------------------------------------
// 4) LSTM recurrence, one persistent workgroup per direction.
//    Thread j owns gate j; h/c/gates live in LDS.
// ---------------------------------------------------------------------------
__global__ __launch_bounds__(1024)
void lstm_kernel(const float* __restrict__ preF, const float* __restrict__ preB,
                 const float* __restrict__ WhhF, const float* __restrict__ WhhB,
                 const float* __restrict__ bihF, const float* __restrict__ bhhF,
                 const float* __restrict__ bihB, const float* __restrict__ bhhB,
                 float* __restrict__ hF, float* __restrict__ hB) {
    __shared__ float h_s[HH_DIM];
    __shared__ float c_s[HH_DIM];
    __shared__ float g_s[G_DIM];

    const int j   = threadIdx.x;
    const int dir = blockIdx.x;
    const float* __restrict__ pre  = dir ? preB : preF;
    const float* __restrict__ Whh  = dir ? WhhB : WhhF;
    float*       __restrict__ hout = dir ? hB   : hF;
    const float bias = dir ? (bihB[j] + bhhB[j]) : (bihF[j] + bhhF[j]);

    if (j < HH_DIM) { h_s[j] = 0.0f; c_s[j] = 0.0f; }
    __syncthreads();

    const float4* __restrict__ wrow = (const float4*)(Whh + (size_t)j * HH_DIM);

    for (int step = 0; step < T_LEN; ++step) {
        const int t = dir ? (T_LEN - 1 - step) : step;

        float acc = pre[(size_t)t * G_DIM + j] + bias;
        const float4* h4 = (const float4*)h_s;
#pragma unroll 8
        for (int k = 0; k < HH_DIM / 4; ++k) {
            float4 w  = wrow[k];
            float4 hv = h4[k];
            acc += w.x * hv.x + w.y * hv.y + w.z * hv.z + w.w * hv.w;
        }
        g_s[j] = acc;
        __syncthreads();

        if (j < HH_DIM) {
            // PyTorch gate order: i, f, g, o
            float ig = sigmoidf_fast(g_s[j]);
            float fg = sigmoidf_fast(g_s[HH_DIM + j]);
            float gg = tanhf(g_s[2 * HH_DIM + j]);
            float og = sigmoidf_fast(g_s[3 * HH_DIM + j]);
            float c  = fg * c_s[j] + ig * gg;
            c_s[j] = c;
            float hv = og * tanhf(c);
            h_s[j] = hv;
            hout[(size_t)t * HH_DIM + j] = hv;
        }
        __syncthreads();
    }
}

// ---------------------------------------------------------------------------
// 5) feats = [hf | hb] @ W_tag^T + b_tag
// ---------------------------------------------------------------------------
__global__ void feats_kernel(const float* __restrict__ hF,
                             const float* __restrict__ hB,
                             const float* __restrict__ Wtag,
                             const float* __restrict__ btag,
                             float* __restrict__ feats) {
    const int t = blockIdx.x;
    const int tag = threadIdx.x;
    if (tag >= NTAGS) return;
    const float* __restrict__ w  = Wtag + (size_t)tag * H_DIM;
    const float* __restrict__ hf = hF + (size_t)t * HH_DIM;
    const float* __restrict__ hb = hB + (size_t)t * HH_DIM;
    float acc = btag[tag];
#pragma unroll 4
    for (int k = 0; k < HH_DIM; ++k) acc += w[k] * hf[k];
#pragma unroll 4
    for (int k = 0; k < HH_DIM; ++k) acc += w[HH_DIM + k] * hb[k];
    feats[t * NTAGS + tag] = acc;
}

// ---------------------------------------------------------------------------
// 6) Viterbi: serial scan (one wave) + single-thread backtrack.
//    out[0] = score, out[1..T] = path tags (as floats).
// ---------------------------------------------------------------------------
__global__ void viterbi_kernel(const float* __restrict__ feats,
                               const float* __restrict__ trans,
                               int* __restrict__ bp,
                               float* __restrict__ out) {
    __shared__ float fv[NTAGS];
    __shared__ float fvn[NTAGS];
    const int j = threadIdx.x;

    if (j < NTAGS) fv[j] = (j == START_TAG) ? 0.0f : NEG_VAL;
    __syncthreads();

    for (int t = 0; t < T_LEN; ++t) {
        if (j < NTAGS) {
            float best = fv[0] + trans[j * NTAGS + 0];
            int bi = 0;
#pragma unroll
            for (int i = 1; i < NTAGS; ++i) {
                float v = fv[i] + trans[j * NTAGS + i];
                if (v > best) { best = v; bi = i; }   // first-max tie break
            }
            bp[t * NTAGS + j] = bi;
            fvn[j] = best + feats[t * NTAGS + j];
        }
        __syncthreads();
        if (j < NTAGS) fv[j] = fvn[j];
        __syncthreads();
    }

    if (j == 0) {
        float best = fv[0] + trans[STOP_TAG * NTAGS + 0];
        int bt = 0;
#pragma unroll
        for (int i = 1; i < NTAGS; ++i) {
            float v = fv[i] + trans[STOP_TAG * NTAGS + i];
            if (v > best) { best = v; bt = i; }
        }
        out[0] = best;
        int tag = bt;
        for (int t = T_LEN - 1; t >= 0; --t) {
            out[1 + t] = (float)tag;
            tag = bp[t * NTAGS + tag];
        }
    }
}

// ---------------------------------------------------------------------------
extern "C" void kernel_launch(void* const* d_in, const int* in_sizes, int n_in,
                              void* d_out, int out_size, void* d_ws, size_t ws_size,
                              hipStream_t stream) {
    const int*   words = (const int*)d_in[0];
    const int*   pos   = (const int*)d_in[1];
    const float* Wword = (const float*)d_in[2];
    const float* Wpos  = (const float*)d_in[3];
    const float* WihF  = (const float*)d_in[4];
    const float* WhhF  = (const float*)d_in[5];
    const float* bihF  = (const float*)d_in[6];
    const float* bhhF  = (const float*)d_in[7];
    const float* WihB  = (const float*)d_in[8];
    const float* WhhB  = (const float*)d_in[9];
    const float* bihB  = (const float*)d_in[10];
    const float* bhhB  = (const float*)d_in[11];
    const float* Wtag  = (const float*)d_in[12];
    const float* btag  = (const float*)d_in[13];
    const float* trans = (const float*)d_in[14];
    float* out = (float*)d_out;

    char* ws = (char*)d_ws;
    bf16_t* Xb    = (bf16_t*)(ws + OFF_XB);
    bf16_t* WbF   = (bf16_t*)(ws + OFF_WBF);
    bf16_t* WbB   = (bf16_t*)(ws + OFF_WBB);
    float*  preF  = (float*)(ws + OFF_PREF);
    float*  preB  = (float*)(ws + OFF_PREB);
    float*  hF    = (float*)(ws + OFF_HF);
    float*  hB    = (float*)(ws + OFF_HB);
    float*  feats = (float*)(ws + OFF_FEATS);
    int*    bpbuf = (int*)(ws + OFF_BP);

    embed_kernel<<<(T_LEN * D_DIM + 255) / 256, 256, 0, stream>>>(
        words, pos, Wword, Wpos, Xb);

    wcvt_kernel<<<(2 * G_DIM * D_DIM + 255) / 256, 256, 0, stream>>>(
        WihF, WihB, WbF, WbB);

    dim3 ggrid(T_LEN / 16, (G_DIM / 16) / 8, 2);     // 256 x 8 x 2
    gemm_kernel<<<ggrid, 256, 0, stream>>>(Xb, WbF, WbB, preF, preB);

    lstm_kernel<<<2, 1024, 0, stream>>>(preF, preB, WhhF, WhhB,
                                        bihF, bhhF, bihB, bhhB, hF, hB);

    feats_kernel<<<T_LEN, 32, 0, stream>>>(hF, hB, Wtag, btag, feats);

    viterbi_kernel<<<1, 32, 0, stream>>>(feats, trans, bpbuf, out);
}
